// L1RegressionMoEActionHead_89876485636873
// MI455X (gfx1250) — matrix operand order repaired and checked by reading, not compiled, not run on hardware
//
#include <hip/hip_runtime.h>
#include <cstdint>

// ---------------------------------------------------------------------------
// MI455X (gfx1250) fused MoE transformer block, v2.
//  * One-time f32->bf16 conversion of activations + selected expert weights.
//  * All GEMMs: v_wmma_f32_16x16x32_bf16, f32 accumulate, 128x128x32 tiles.
//  * K-tile staging via Tensor Data Mover (tensor_load_to_lds, TENSORcnt),
//    double-buffered so DMA overlaps WMMA. TDM padding gives the 48-elem
//    LDS row stride (bank-conflict avoidance) for free.
//  * Attention: per (b,h,64-query block); Q/K tiles via TDM; scores f32 in
//    LDS; P and V^T bf16; both attention GEMMs on WMMA.
// ---------------------------------------------------------------------------

typedef __bf16 v16bf __attribute__((ext_vector_type(16)));
typedef float  v8f   __attribute__((ext_vector_type(8)));
typedef unsigned int v4u  __attribute__((ext_vector_type(4)));
typedef int          v8i_ __attribute__((ext_vector_type(8)));
typedef int          v4i_ __attribute__((ext_vector_type(4)));

#define DIM 1024

// f32 -> bf16 RTNE, integer-only.
__device__ inline unsigned short f2bf(float f) {
    unsigned u = __float_as_uint(f);
    u += 0x7FFFu + ((u >> 16) & 1u);
    return (unsigned short)(u >> 16);
}

// Convert 16 f32 -> 16 bf16, store as two uint4 (works for LDS or global dst).
__device__ inline void cvt_store16(const float* __restrict__ g, unsigned short* __restrict__ dst) {
    float4 f0 = reinterpret_cast<const float4*>(g)[0];
    float4 f1 = reinterpret_cast<const float4*>(g)[1];
    float4 f2 = reinterpret_cast<const float4*>(g)[2];
    float4 f3 = reinterpret_cast<const float4*>(g)[3];
    union { unsigned short h[16]; uint4 u[2]; } un;
    const float fs[16] = {f0.x,f0.y,f0.z,f0.w, f1.x,f1.y,f1.z,f1.w,
                          f2.x,f2.y,f2.z,f2.w, f3.x,f3.y,f3.z,f3.w};
#pragma unroll
    for (int i = 0; i < 16; ++i) un.h[i] = f2bf(fs[i]);
    reinterpret_cast<uint4*>(dst)[0] = un.u[0];
    reinterpret_cast<uint4*>(dst)[1] = un.u[1];
}

// A-matrix 16x32 bf16 fragment (ISA 7.12.2 layout).
__device__ inline v16bf load_fragA(const unsigned short* base, int stride) {
    int lane = threadIdx.x & 31;
    const unsigned short* p = base + (lane & 15) * stride + ((lane >> 4) << 3);
    union { v16bf v; uint4 u[2]; } un;
    un.u[0] = *reinterpret_cast<const uint4*>(p);
    un.u[1] = *reinterpret_cast<const uint4*>(p + 16);
    return un.v;
}

// B-matrix 32x16 bf16 fragment: lane = N, K in ISA table order (8h+s, 16+8h+s-8).
__device__ inline v16bf load_fragB(const unsigned short* base, int stride) {
    int lane = threadIdx.x & 31;
    const unsigned short* p = base + (lane & 15) * stride + ((lane >> 4) << 3);
    union { v16bf v; uint4 u[2]; } un;
    un.u[0] = *reinterpret_cast<const uint4*>(p);
    un.u[1] = *reinterpret_cast<const uint4*>(p + 16);
    return un.v;
}

// ---------------------------------------------------------------------------
// TDM 2D tile load: bf16 tensor (data_size=2B), tileRows x tileK tile,
// row stride strideK elems, LDS destination at byte offset lds_off with
// optional padding (pad codes per D# group1 encoding). Single descriptor.
// Issued by one wave; completion via s_wait_tensorcnt.
// ---------------------------------------------------------------------------
__device__ inline void tdm_load_2d(unsigned lds_off, const void* gptr,
                                   unsigned tileK, unsigned tileRows,
                                   unsigned tensorK, unsigned tensorRows,
                                   unsigned strideK,
                                   int padIntervalCode, int padAmountCode)
{
    unsigned long long ga = (unsigned long long)(uintptr_t)gptr;
    v4u g0 = { 1u,                                        // count=1, user mode
               lds_off,                                   // lds_addr
               (unsigned)(ga & 0xFFFFFFFFu),              // global_addr[31:0]
               (unsigned)((ga >> 32) & 0x01FFFFFFu) | (2u << 30) }; // [56:32] | type=2
    v8i_ g1;
    g1[0] = (int)((1u << 16)                               // data_size = 2B
                | (1u << 20)                               // pad_enable
                | ((unsigned)padIntervalCode << 22)
                | ((unsigned)padAmountCode << 25));
    g1[1] = (int)(tensorK << 16);                          // tensor_dim0[15:0]
    g1[2] = (int)(((tensorK >> 16) & 0xFFFFu) | (tensorRows << 16));
    g1[3] = (int)(((tensorRows >> 16) & 0xFFFFu) | (tileK << 16));
    g1[4] = (int)(tileRows & 0xFFFFu);                     // tile_dim1 (tile_dim2=0)
    g1[5] = (int)strideK;                                  // tensor_dim0_stride[31:0]
    g1[6] = 0;
    g1[7] = 0;
    v4i_ z = {0, 0, 0, 0};
#if __clang_major__ >= 23
    v8i_ z8 = {0, 0, 0, 0, 0, 0, 0, 0};
    __builtin_amdgcn_tensor_load_to_lds(g0, g1, z, z, z8, 0);
#else
    __builtin_amdgcn_tensor_load_to_lds(g0, g1, z, z, 0);
#endif
}

// ---------------------------------------------------------------------------
// f32 -> bf16 streaming converter; optional expert select (src += e*estride).
// 16 elems per thread, grid sized exactly.
// ---------------------------------------------------------------------------
__global__ __launch_bounds__(256)
void cvt_bf16_kernel(const float* __restrict__ src, const int* __restrict__ eidx,
                     unsigned long long estride, unsigned short* __restrict__ dst)
{
    const size_t base = ((size_t)blockIdx.x * 256 + threadIdx.x) * 16;
    cvt_store16(src + (size_t)eidx[0] * estride + base, dst + base);
}

// ---------------------------------------------------------------------------
// C[M,1024] = A[M,1024](bf16) @ W[1024,1024]^T(bf16) + b[e]  (+RoPE/+res/+ReLU)
// grid = (8, M/128), block 256 (8 waves). TDM double-buffered K staging.
// LDS: A0 @0, A1 @12288, B0 @24576, B1 @36864 (each 128 x 48-stride bf16).
// ---------------------------------------------------------------------------
template<int ROPE_LEN, bool DO_RELU, bool DO_RES, bool OUT_BF16>
__global__ __launch_bounds__(256)
void gemm_bf16_kernel(const unsigned short* __restrict__ A,
                      const unsigned short* __restrict__ W,
                      const float* __restrict__ bias,
                      const int* __restrict__ eidx,
                      const float* __restrict__ resid,
                      void* __restrict__ Cout)
{
    __shared__ char lds[4 * 12288];
    const int tid = threadIdx.x, lane = tid & 31, wave = tid >> 5;
    const int e = eidx[0];
    const float* be = bias + (size_t)e * DIM;
    const int mBase = blockIdx.y * 128, nBase = blockIdx.x * 128;
    const unsigned M = gridDim.y * 128u;
    const int wm = wave >> 2, wn = wave & 3;

    // Prologue: K-tile 0 into buffer 0. pad codes: 16 DW interval, 8 DW pad
    // => LDS row stride 48 bf16 (96B, 16B aligned, conflict-staggered).
    if (wave == 0) {
        tdm_load_2d(0u,      A + (size_t)mBase * DIM, 32, 128, DIM, M,    DIM, 3, 7);
        tdm_load_2d(24576u,  W + (size_t)nBase * DIM, 32, 128, DIM, DIM,  DIM, 3, 7);
        __builtin_amdgcn_s_wait_tensorcnt(0);
    }
    __syncthreads();

    v8f acc[4][2] = {};
    for (int t = 0; t < DIM / 32; ++t) {
        const unsigned cur = (unsigned)t & 1u;
        if (wave == 0 && t + 1 < DIM / 32) {   // prefetch next K tile (overlaps WMMA)
            const unsigned nb = (t + 1) & 1;
            const int k1 = (t + 1) * 32;
            tdm_load_2d(nb * 12288u,          A + (size_t)mBase * DIM + k1, 32, 128, DIM, M,   DIM, 3, 7);
            tdm_load_2d(24576u + nb * 12288u, W + (size_t)nBase * DIM + k1, 32, 128, DIM, DIM, DIM, 3, 7);
        }
        const unsigned short* Abuf = (const unsigned short*)(lds + cur * 12288u);
        const unsigned short* Bbuf = (const unsigned short*)(lds + 24576u + cur * 12288u);
        v16bf bf0 = load_fragB(Bbuf + (wn * 32 +  0) * 48, 48);
        v16bf bf1 = load_fragB(Bbuf + (wn * 32 + 16) * 48, 48);
#pragma unroll
        for (int mt = 0; mt < 4; ++mt) {
            v16bf af = load_fragA(Abuf + (wm * 64 + mt * 16) * 48, 48);
            acc[mt][0] = __builtin_amdgcn_wmma_f32_16x16x32_bf16(false, af, false, bf0,
                             (short)0, acc[mt][0], false, false);
            acc[mt][1] = __builtin_amdgcn_wmma_f32_16x16x32_bf16(false, af, false, bf1,
                             (short)0, acc[mt][1], false, false);
        }
        __syncthreads();   // all waves done reading buffer `cur`
        if (wave == 0 && t + 1 < DIM / 32) __builtin_amdgcn_s_wait_tensorcnt(0);
        __syncthreads();   // next buffer visible to all waves
    }

    const int rowOff = (lane >> 4) << 3, nLane = lane & 15;
#pragma unroll
    for (int mt = 0; mt < 4; ++mt) {
#pragma unroll
        for (int nt = 0; nt < 2; ++nt) {
            const int col = nBase + wn * 32 + nt * 16 + nLane;
            const float bv = be[col];
#pragma unroll
            for (int j = 0; j < 8; ++j) {
                const int row = mBase + wm * 64 + mt * 16 + rowOff + j;
                float v = acc[mt][nt][j] + bv;
                if constexpr (DO_RES) v += resid[(size_t)row * DIM + col];
                if constexpr (ROPE_LEN > 0) {
                    float other = __shfl_xor(v, 1, 32);   // rotate-half neighbor
                    float rot = (col & 1) ? other : -other;
                    int i2 = col & 31;
                    float invf = __expf(-(float)i2 * (9.210340371976184f / 32.f));
                    float ang = (float)(row % ROPE_LEN) * invf;
                    float sn, cs; __sincosf(ang, &sn, &cs);
                    v = v * cs + rot * sn;
                }
                if constexpr (DO_RELU) v = fmaxf(v, 0.f);
                const size_t idx = (size_t)row * DIM + col;
                if constexpr (OUT_BF16) ((unsigned short*)Cout)[idx] = f2bf(v);
                else                    ((float*)Cout)[idx] = v;
            }
        }
    }
}

// ---------------------------------------------------------------------------
// Attention. grid (8, NH=8, B=16), block 256. Dynamic LDS (starts at 0):
//   sc @0       : 64x513 f32  (131328 B)
//   Pb @131328  : 64x520 bf16 ( 66560 B)
//   Kt @197888  : 256x72 bf16 ( 36864 B)   <- TDM target
//   Qb @234752  : 64x72  bf16 (  9216 B)   <- TDM target
//   Vt @243968  : 64x520 bf16 ( 66560 B)
// total 310528 B <= 320KB.
// ---------------------------------------------------------------------------
#define ATTN_SMEM 310528
#define OFF_P 131328
#define OFF_K 197888
#define OFF_Q 234752
#define OFF_V 243968

__global__ __launch_bounds__(256)
void attn_kernel(const unsigned short* __restrict__ qa, const unsigned short* __restrict__ qt_,
                 const unsigned short* __restrict__ ka, const unsigned short* __restrict__ va,
                 const unsigned short* __restrict__ kt_, const unsigned short* __restrict__ vt_,
                 const float* __restrict__ gate, const int* __restrict__ eidx,
                 unsigned short* __restrict__ out)
{
    extern __shared__ char smem[];
    float*          sc = (float*)smem;
    unsigned short* Pb = (unsigned short*)(smem + OFF_P);
    unsigned short* Kt = (unsigned short*)(smem + OFF_K);
    unsigned short* Qb = (unsigned short*)(smem + OFF_Q);
    unsigned short* Vt = (unsigned short*)(smem + OFF_V);

    const int tid = threadIdx.x, lane = tid & 31, wave = tid >> 5;
    const int t0 = blockIdx.x * 64, h = blockIdx.y, b = blockIdx.z;
    const int e = eidx[0];
    const float ratio = 1.f / (1.f + __expf(-gate[e]));

    // Stage V^T (both memories) bf16: Vt[d][k], k in [0,512)
#pragma unroll
    for (int i = 0; i < 8; ++i) {
        int c  = tid + 256 * i;
        int kk = c >> 2;
        int db = (c & 3) * 16;
        const unsigned short* src = ((kk < 256) ? va : vt_)
                                    + (size_t)(b * 256 + (kk & 255)) * DIM + h * 64 + db;
#pragma unroll
        for (int j = 0; j < 16; ++j) Vt[(db + j) * 520 + kk] = src[j];
    }

    for (int mem = 0; mem < 2; ++mem) {
        __syncthreads();
        if (wave == 0) {   // TDM: Q 64x64 and K 256x64 tiles, pad to 72-elem stride
            const unsigned short* qsrc = (mem ? qt_ : qa) + (size_t)(b * 512 + t0) * DIM + h * 64;
            const unsigned short* ksrc = (mem ? kt_ : ka) + (size_t)(b * 256) * DIM + h * 64;
            tdm_load_2d(OFF_Q, qsrc, 64, 64,  DIM, 64u * 1024u, DIM, 4, 3);
            tdm_load_2d(OFF_K, ksrc, 64, 256, DIM, 64u * 1024u, DIM, 4, 3);
            __builtin_amdgcn_s_wait_tensorcnt(0);
        }
        __syncthreads();
        const float scl = mem ? 0.125f * ratio : 0.125f;
#pragma unroll
        for (int i = 0; i < 8; ++i) {   // 64 score tiles / 8 waves
            int job = wave * 8 + i;
            int mt = job >> 4, nt = job & 15;
            v16bf a0 = load_fragA(Qb + (mt * 16) * 72 +  0, 72);
            v16bf a1 = load_fragA(Qb + (mt * 16) * 72 + 32, 72);
            v16bf b0 = load_fragB(Kt + (nt * 16) * 72 +  0, 72);
            v16bf b1 = load_fragB(Kt + (nt * 16) * 72 + 32, 72);
            v8f s = {};
            s = __builtin_amdgcn_wmma_f32_16x16x32_bf16(false, a0, false, b0, (short)0, s, false, false);
            s = __builtin_amdgcn_wmma_f32_16x16x32_bf16(false, a1, false, b1, (short)0, s, false, false);
            int m0 = mt * 16 + ((lane >> 4) << 3);
            int n  = mem * 256 + nt * 16 + (lane & 15);
#pragma unroll
            for (int j = 0; j < 8; ++j) sc[(m0 + j) * 513 + n] = s[j] * scl;
        }
    }
    __syncthreads();

    if (tid < 64) {   // softmax over 512, one row per thread
        float* row = sc + tid * 513;
        float mx = -1e30f;
        for (int i = 0; i < 512; ++i) mx = fmaxf(mx, row[i]);
        float sum = 0.f;
        for (int i = 0; i < 512; ++i) { float ev = __expf(row[i] - mx); row[i] = ev; sum += ev; }
        float inv = 1.f / sum;
        for (int i = 0; i < 512; ++i) Pb[tid * 520 + i] = f2bf(row[i] * inv);
    }
    __syncthreads();

    // out(64x64) = P(64x512) @ V(512x64); 16 tiles / 8 waves
#pragma unroll
    for (int i = 0; i < 2; ++i) {
        int t_ = wave * 2 + i;
        int mt = t_ >> 2, nt = t_ & 3;
        v8f o = {};
        for (int k0 = 0; k0 < 512; k0 += 32) {
            v16bf a  = load_fragA(Pb + (mt * 16) * 520 + k0, 520);
            v16bf bb = load_fragB(Vt + (nt * 16) * 520 + k0, 520);
            o = __builtin_amdgcn_wmma_f32_16x16x32_bf16(false, a, false, bb, (short)0, o, false, false);
        }
        int m0 = mt * 16 + ((lane >> 4) << 3);
        int d  = nt * 16 + (lane & 15);
#pragma unroll
        for (int j = 0; j < 8; ++j)
            out[(size_t)(b * 512 + t0 + m0 + j) * DIM + h * 64 + d] = f2bf(o[j]);
    }
}

// ---------------------------------------------------------------------------
// LayerNorm: one wave32 per row; f32 in, bf16 out (feeds FFN GEMM).
// ---------------------------------------------------------------------------
__global__ __launch_bounds__(256)
void ln_kernel(const float* __restrict__ X, const float* __restrict__ gamma,
               const float* __restrict__ beta, const int* __restrict__ eidx,
               unsigned short* __restrict__ Y)
{
    const int lane = threadIdx.x & 31, wv = threadIdx.x >> 5;
    const int row = blockIdx.x * 8 + wv;
    const int e = eidx[0];
    const float* xr = X + (size_t)row * DIM;
    const float* ge = gamma + (size_t)e * DIM;
    const float* be = beta  + (size_t)e * DIM;
    float s = 0.f, s2 = 0.f;
    for (int i = lane; i < DIM; i += 32) { float v = xr[i]; s += v; s2 += v * v; }
#pragma unroll
    for (int o = 16; o > 0; o >>= 1) { s += __shfl_xor(s, o, 32); s2 += __shfl_xor(s2, o, 32); }
    float mu  = s * (1.f / DIM);
    float var = s2 * (1.f / DIM) - mu * mu;
    float rs  = rsqrtf(var + 1e-5f);
    for (int i = lane; i < DIM; i += 32)
        Y[(size_t)row * DIM + i] = f2bf((xr[i] - mu) * rs * ge[i] + be[i]);
}

// ---------------------------------------------------------------------------
extern "C" void kernel_launch(void* const* d_in, const int* in_sizes, int n_in,
                              void* d_out, int out_size, void* d_ws, size_t ws_size,
                              hipStream_t stream) {
    const float* x    = (const float*)d_in[0];
    const float* h_a  = (const float*)d_in[1];
    const float* h_t  = (const float*)d_in[2];
    const float* W_qa = (const float*)d_in[3];  const float* b_qa = (const float*)d_in[4];
    const float* W_ka = (const float*)d_in[5];  const float* b_ka = (const float*)d_in[6];
    const float* W_va = (const float*)d_in[7];  const float* b_va = (const float*)d_in[8];
    const float* W_qt = (const float*)d_in[9];  const float* b_qt = (const float*)d_in[10];
    const float* W_kt = (const float*)d_in[11]; const float* b_kt = (const float*)d_in[12];
    const float* W_vt = (const float*)d_in[13]; const float* b_vt = (const float*)d_in[14];
    const float* W_o  = (const float*)d_in[15]; const float* b_o  = (const float*)d_in[16];
    const float* W_ff = (const float*)d_in[17]; const float* b_ff = (const float*)d_in[18];
    const float* gamma = (const float*)d_in[19];
    const float* beta  = (const float*)d_in[20];
    const float* gate  = (const float*)d_in[21];
    const int*   eidx  = (const int*)d_in[22];
    float* out = (float*)d_out;

    // Workspace (bytes). Peak 112 MB with overlays.
    char* wsb = (char*)d_ws;
    const size_t MB = 1024ull * 1024ull;
    unsigned short* xb    = (unsigned short*)(wsb + 0);        // 16MB (reused: attnb)
    unsigned short* hab   = (unsigned short*)(wsb + 16 * MB);  //  8MB (reused: xnb)
    unsigned short* htb   = (unsigned short*)(wsb + 24 * MB);  //  8MB
    unsigned short* Wb    = (unsigned short*)(wsb + 32 * MB);  // 16MB: 8 x 1Mi bf16
    unsigned short* qab   = (unsigned short*)(wsb + 48 * MB);  // 16MB (oln overlays 48..80MB)
    unsigned short* qtb   = (unsigned short*)(wsb + 64 * MB);  // 16MB
    unsigned short* kab   = (unsigned short*)(wsb + 80 * MB);  //  8MB
    unsigned short* vab   = (unsigned short*)(wsb + 88 * MB);  //  8MB
    unsigned short* ktb   = (unsigned short*)(wsb + 96 * MB);  //  8MB
    unsigned short* vtb   = (unsigned short*)(wsb + 104 * MB); //  8MB
    unsigned short* attnb = xb;
    float*          oln   = (float*)(wsb + 48 * MB);
    unsigned short* xnb   = hab;
    const size_t MM = (size_t)DIM * DIM;   // elems per weight matrix

    dim3 blk(256);
    dim3 gX(8, 64);   // M = 8192
    dim3 gH(8, 32);   // M = 4096

    // One-time f32 -> bf16 conversions (activations + selected expert weights)
    cvt_bf16_kernel<<<2048, blk, 0, stream>>>(x,   eidx, 0ull, xb);
    cvt_bf16_kernel<<<1024, blk, 0, stream>>>(h_a, eidx, 0ull, hab);
    cvt_bf16_kernel<<<1024, blk, 0, stream>>>(h_t, eidx, 0ull, htb);
    const float* Wsrc[8] = {W_qa, W_ka, W_va, W_qt, W_kt, W_vt, W_o, W_ff};
    for (int i = 0; i < 8; ++i)
        cvt_bf16_kernel<<<256, blk, 0, stream>>>(Wsrc[i], eidx, (unsigned long long)MM,
                                                 Wb + (size_t)i * MM);

    // Projections (+bias, +RoPE on q/k) -> bf16
    gemm_bf16_kernel<512, false, false, true><<<gX, blk, 0, stream>>>(xb,  Wb + 0 * MM, b_qa, eidx, nullptr, qab);
    gemm_bf16_kernel<512, false, false, true><<<gX, blk, 0, stream>>>(xb,  Wb + 3 * MM, b_qt, eidx, nullptr, qtb);
    gemm_bf16_kernel<256, false, false, true><<<gH, blk, 0, stream>>>(hab, Wb + 1 * MM, b_ka, eidx, nullptr, kab);
    gemm_bf16_kernel<0,   false, false, true><<<gH, blk, 0, stream>>>(hab, Wb + 2 * MM, b_va, eidx, nullptr, vab);
    gemm_bf16_kernel<256, false, false, true><<<gH, blk, 0, stream>>>(htb, Wb + 4 * MM, b_kt, eidx, nullptr, ktb);
    gemm_bf16_kernel<0,   false, false, true><<<gH, blk, 0, stream>>>(htb, Wb + 5 * MM, b_vt, eidx, nullptr, vtb);

    // Dual-memory attention
    hipFuncSetAttribute((const void*)attn_kernel,
                        hipFuncAttributeMaxDynamicSharedMemorySize, ATTN_SMEM);
    attn_kernel<<<dim3(8, 8, 16), blk, ATTN_SMEM, stream>>>(qab, qtb, kab, vab, ktb, vtb,
                                                            gate, eidx, attnb);

    // O-projection + residual (f32 out), LayerNorm (bf16 out), FFN + ReLU (f32 out)
    gemm_bf16_kernel<0, false, true,  false><<<gX, blk, 0, stream>>>(attnb, Wb + 6 * MM, b_o,  eidx, x, oln);
    ln_kernel<<<1024, blk, 0, stream>>>(oln, gamma, beta, eidx, xnb);
    gemm_bf16_kernel<0, true,  false, false><<<gX, blk, 0, stream>>>(xnb,   Wb + 7 * MM, b_ff, eidx, nullptr, out);
}
